// AdaCOFModel_81088982548606
// MI455X (gfx1250) — matrix operand order repaired and checked by reading, not compiled, not run on hardware
//
#include <hip/hip_runtime.h>
#include <hip/hip_bf16.h>
#include <cstdint>
#include <cstddef>

// ---------------------------------------------------------------------------
// AdaCOF-style frame interpolation for MI455X (gfx1250).
// All 3x3 convolutions run as implicit GEMM on v_wmma_f32_16x16x32_f16.
//
// Round-3 changes (from compile feedback):
//  * 2x2 register blocking in the conv: each wave computes two 16-row M tiles
//    x two 16-pixel N tiles (4 independent WMMA accumulator chains). Each
//    (tap,chunk) step issues 2 A-fragment + 2 B-fragment loads for 4 WMMAs
//    -> 2 b128 loads per WMMA instead of 3.
//  * Keeps round-2 wins: 1-px zero halo (no bounds checks / cndmask in the
//    K loop), NHWC fp16 with channels padded to multiples of 32 (B fragment =
//    single contiguous 32B load), weights repacked to A-fragment lane order.
// Workspace peak ~750 MB (dominated by the 512-ch fp16 intermediate).
// ---------------------------------------------------------------------------

typedef __attribute__((ext_vector_type(16))) _Float16 v16h;
typedef __attribute__((ext_vector_type(8)))  _Float16 v8h;
typedef __attribute__((ext_vector_type(8)))  float    v8f;

#define NB   2
#define HH   384
#define WW   384
#define HP   (HH + 2)          // padded height (1-px zero halo)
#define WP   (WW + 2)          // padded width
#define HWSZ (HH * WW)
#define PIX  (NB * HH * WW)    // 294912 interior pixels, divisible by 32
#define PIXP (NB * HP * WP)    // padded pixel count

static __device__ __forceinline__ size_t pad_off(int b, int rem)
{
    int y = rem / WW, x = rem % WW;
    return ((size_t)b * HP + (y + 1)) * WP + (x + 1);
}

// ---------------------------------------------------------------------------
// Zero the spatial halo of a padded NHWC buffer (interior is fully written by
// the producing conv, so this is the only initialization needed per launch).
// ---------------------------------------------------------------------------
__global__ __launch_bounds__(256) void halo_zero_kernel(
    _Float16* __restrict__ buf, int Cpad)
{
    int idx = blockIdx.x * 256 + threadIdx.x;
    if (idx >= PIXP) return;
    int rem = idx % (HP * WP);
    int y = rem / WP, x = rem % WP;
    if (y == 0 || y == HP - 1 || x == 0 || x == WP - 1) {
        _Float16* o = buf + (size_t)idx * Cpad;
        for (int c = 0; c < Cpad; ++c) o[c] = (_Float16)0.0f;
    }
}

// ---------------------------------------------------------------------------
// Repack (Cout, CinSrc, 3, 3) fp32 weights into WMMA A-fragment order:
// packed[((mt*9 + t)*nch + c)*32 + lane][16 halves].
// A order used here (contiguous k, same as B): lanes 0-15 hold K = {0..15},
// lanes 16-31 hold K = {16..31}; M = lane & 15.
// dup!=0 implements conv(concat(f,f), W): weight(m,cin) = W[m,cin] + W[m,cin+Cin].
// ---------------------------------------------------------------------------
__global__ __launch_bounds__(256) void pack_w_kernel(
    const float* __restrict__ w, _Float16* __restrict__ wp,
    int Cout, int Cin, int CinSrc, int nch, int dup, int total)
{
    int idx = blockIdx.x * 256 + threadIdx.x;
    if (idx >= total) return;
    int j    = idx & 15;
    int l    = (idx >> 4) & 31;
    int rest = idx >> 9;                 // (mt*9 + t)*nch + c
    int c    = rest % nch;
    int t    = (rest / nch) % 9;
    int mt   = rest / (nch * 9);
    int K    = j + ((l < 16) ? 0 : 16);
    int m    = mt * 16 + (l & 15);
    int cin  = c * 32 + K;
    float v = 0.0f;
    if (m < Cout && cin < Cin) {
        size_t base = (((size_t)m * CinSrc + cin) * 3 + t / 3) * 3 + (t % 3);
        v = w[base];
        if (dup) v += w[base + (size_t)Cin * 9];
    }
    wp[idx] = (_Float16)v;
}

// ---------------------------------------------------------------------------
// 3x3 SAME conv, implicit GEMM on WMMA with 2x2 register blocking:
// each wave computes M tiles {2mp, 2mp+1} x N tiles {2np, 2np+1}
// (32 output channels x 32 pixels). 4 WMMAs per 4 fragment loads.
// act/out: padded NHWC fp16 (zero halo). No branches in the K loop.
// ---------------------------------------------------------------------------
__global__ __launch_bounds__(256) void conv3x3_wmma_kernel(
    const _Float16* __restrict__ act, const _Float16* __restrict__ wpack,
    const float* __restrict__ bias, _Float16* __restrict__ outp,
    int CinPad, int nch, int Cout, int CoutPad, int Mpairs,
    int relu_flag, float scale)
{
    int wave = threadIdx.x >> 5;
    int lane = threadIdx.x & 31;
    int tid  = blockIdx.x * 8 + wave;              // tile id (wave-uniform)
    int tiles = Mpairs * (PIX / 32);
    if (tid >= tiles) return;

    int mp = tid % Mpairs;                         // waves in a block share pixels
    int np = tid / Mpairs;
    int n  = lane & 15;
    int kb = (lane < 16) ? 0 : 16;                 // B-fragment K base

    int pix0 = np * 32 + n;                        // N tile 0 pixel of this lane
    int pix1 = pix0 + 16;                          // N tile 1 pixel of this lane
    int b0 = pix0 / HWSZ, r0 = pix0 % HWSZ, y0 = r0 / WW, x0 = r0 % WW;
    int b1 = pix1 / HWSZ, r1 = pix1 % HWSZ, y1 = r1 / WW, x1 = r1 % WW;

    v8f acc00 = {}, acc01 = {}, acc10 = {}, acc11 = {};
    size_t wstride = (size_t)9 * nch * 512;        // halves per M tile
    const _Float16* wp0 = wpack + (size_t)(2 * mp) * wstride + (size_t)lane * 16;
    const _Float16* wp1 = wp0 + wstride;

    #pragma unroll
    for (int t = 0; t < 9; ++t) {
        // padded coords: (y + dy + 1) = y + t/3, (x + dx + 1) = x + t%3
        const _Float16* bp0 =
            act + (((size_t)b0 * HP + y0 + t / 3) * WP + x0 + t % 3) * CinPad + kb;
        const _Float16* bp1 =
            act + (((size_t)b1 * HP + y1 + t / 3) * WP + x1 + t % 3) * CinPad + kb;
        for (int c = 0; c < nch; ++c) {
            v16h af0 = *(const v16h*)wp0; wp0 += 512;   // A tile 0 (32B)
            v16h af1 = *(const v16h*)wp1; wp1 += 512;   // A tile 1 (32B)
            v16h bf0 = *(const v16h*)(bp0 + c * 32);    // B tile 0 (32B)
            v16h bf1 = *(const v16h*)(bp1 + c * 32);    // B tile 1 (32B)
            acc00 = __builtin_amdgcn_wmma_f32_16x16x32_f16(
                false, af0, false, bf0, (short)0, acc00, false, false);
            acc10 = __builtin_amdgcn_wmma_f32_16x16x32_f16(
                false, af1, false, bf0, (short)0, acc10, false, false);
            acc01 = __builtin_amdgcn_wmma_f32_16x16x32_f16(
                false, af0, false, bf1, (short)0, acc01, false, false);
            acc11 = __builtin_amdgcn_wmma_f32_16x16x32_f16(
                false, af1, false, bf1, (short)0, acc11, false, false);
        }
    }

    // Epilogue: bias + optional ReLU + scale; 8 contiguous channels per store.
    int lhalf = (lane < 16) ? 0 : 8;
    int mb0 = mp * 32 + lhalf;
    int mb1 = mp * 32 + 16 + lhalf;
    float bia0[8], bia1[8];
    #pragma unroll
    for (int r = 0; r < 8; ++r) {
        bia0[r] = (mb0 + r) < Cout ? bias[mb0 + r] : 0.0f;
        bia1[r] = (mb1 + r) < Cout ? bias[mb1 + r] : 0.0f;
    }
    size_t ob0 = (((size_t)b0 * HP + y0 + 1) * WP + x0 + 1) * CoutPad;
    size_t ob1 = (((size_t)b1 * HP + y1 + 1) * WP + x1 + 1) * CoutPad;
    v8h o00, o01, o10, o11;
    #pragma unroll
    for (int r = 0; r < 8; ++r) {
        float v00 = acc00[r] + bia0[r], v10 = acc10[r] + bia1[r];
        float v01 = acc01[r] + bia0[r], v11 = acc11[r] + bia1[r];
        if (relu_flag) {
            v00 = fmaxf(v00, 0.0f); v10 = fmaxf(v10, 0.0f);
            v01 = fmaxf(v01, 0.0f); v11 = fmaxf(v11, 0.0f);
        }
        o00[r] = (_Float16)(v00 * scale); o10[r] = (_Float16)(v10 * scale);
        o01[r] = (_Float16)(v01 * scale); o11[r] = (_Float16)(v11 * scale);
    }
    *(v8h*)(outp + ob0 + mb0) = o00;
    *(v8h*)(outp + ob0 + mb1) = o10;
    *(v8h*)(outp + ob1 + mb0) = o01;
    *(v8h*)(outp + ob1 + mb1) = o11;
}

// ---------------------------------------------------------------------------
// Assemble the padded NHWC fp16 network input (32 padded channels):
// ch0-2 = frame1, ch3-5 = frame2, [ch6-13 = 4 flows if given], rest zero.
// ---------------------------------------------------------------------------
__global__ __launch_bounds__(256) void assemble_kernel(
    const float* __restrict__ f1, const float* __restrict__ f2,
    const _Float16* __restrict__ flows, _Float16* __restrict__ xo)
{
    int pix = blockIdx.x * 256 + threadIdx.x;
    if (pix >= PIX) return;
    int b = pix / HWSZ, rem = pix % HWSZ;
    size_t po = pad_off(b, rem);
    _Float16* o = xo + po * 32;
    #pragma unroll
    for (int c = 0; c < 3; ++c) {
        o[c]     = (_Float16)f1[(size_t)(b * 3 + c) * HWSZ + rem];
        o[3 + c] = (_Float16)f2[(size_t)(b * 3 + c) * HWSZ + rem];
    }
    if (flows) {
        #pragma unroll
        for (int f = 0; f < 4; ++f) {
            const _Float16* fl = flows + ((size_t)f * PIXP + po) * 32;
            o[6 + 2 * f] = fl[0];
            o[7 + 2 * f] = fl[1];
        }
        #pragma unroll
        for (int c = 14; c < 32; ++c) o[c] = (_Float16)0.0f;
    } else {
        #pragma unroll
        for (int c = 6; c < 32; ++c) o[c] = (_Float16)0.0f;
    }
}

// ---------------------------------------------------------------------------
// Bilinear warp (border clamp, align_corners) + blend. T = 0.5 so the
// backward warp of frame2 uses exactly -flow.  blended: (F, B, 3, H, W) fp32.
// ---------------------------------------------------------------------------
static __device__ __forceinline__ int imin(int a, int b) { return a < b ? a : b; }

static __device__ __forceinline__ float bilin(const float* img, float px, float py)
{
    int x0 = (int)floorf(px), y0 = (int)floorf(py);
    int x1 = imin(x0 + 1, WW - 1), y1 = imin(y0 + 1, HH - 1);
    float wx = px - (float)x0, wy = py - (float)y0;
    float v00 = img[y0 * WW + x0], v01 = img[y0 * WW + x1];
    float v10 = img[y1 * WW + x0], v11 = img[y1 * WW + x1];
    return v00 * (1.f - wx) * (1.f - wy) + v01 * wx * (1.f - wy)
         + v10 * (1.f - wx) * wy + v11 * wx * wy;
}

__global__ __launch_bounds__(256) void warp_blend_kernel(
    const float* __restrict__ f1, const float* __restrict__ f2,
    const _Float16* __restrict__ flows, float* __restrict__ blended)
{
    int idx = blockIdx.x * 256 + threadIdx.x;
    if (idx >= 4 * PIX) return;
    int f = idx / PIX, pix = idx % PIX;
    int b = pix / HWSZ, rem = pix % HWSZ;
    int y = rem / WW, x = rem % WW;
    const _Float16* fl = flows + ((size_t)f * PIXP + pad_off(b, rem)) * 32;
    float u = (float)fl[0], v = (float)fl[1];
    float px1 = fminf(fmaxf((float)x + u, 0.f), (float)(WW - 1));
    float py1 = fminf(fmaxf((float)y + v, 0.f), (float)(HH - 1));
    float px2 = fminf(fmaxf((float)x - u, 0.f), (float)(WW - 1));
    float py2 = fminf(fmaxf((float)y - v, 0.f), (float)(HH - 1));
    #pragma unroll
    for (int c = 0; c < 3; ++c) {
        const float* i1 = f1 + (size_t)(b * 3 + c) * HWSZ;
        const float* i2 = f2 + (size_t)(b * 3 + c) * HWSZ;
        float w1 = bilin(i1, px1, py1);
        float w2 = bilin(i2, px2, py2);
        blended[((size_t)(f * NB + b) * 3 + c) * HWSZ + rem] = 0.5f * (w1 + w2);
    }
}

// ---------------------------------------------------------------------------
// Softmax over F=4 collaboration weights -> cw output (B, 4, H, W) fp32.
// ---------------------------------------------------------------------------
__global__ __launch_bounds__(256) void cw_softmax_kernel(
    const _Float16* __restrict__ co, float* __restrict__ cw)
{
    int pix = blockIdx.x * 256 + threadIdx.x;
    if (pix >= PIX) return;
    int b = pix / HWSZ, rem = pix % HWSZ;
    const _Float16* c = co + pad_off(b, rem) * 32;
    float v0 = (float)c[0], v1 = (float)c[1], v2 = (float)c[2], v3 = (float)c[3];
    float mx = fmaxf(fmaxf(v0, v1), fmaxf(v2, v3));
    float e0 = __expf(v0 - mx), e1 = __expf(v1 - mx);
    float e2 = __expf(v2 - mx), e3 = __expf(v3 - mx);
    float inv = 1.0f / (e0 + e1 + e2 + e3);
    cw[((size_t)b * 4 + 0) * HWSZ + rem] = e0 * inv;
    cw[((size_t)b * 4 + 1) * HWSZ + rem] = e1 * inv;
    cw[((size_t)b * 4 + 2) * HWSZ + rem] = e2 * inv;
    cw[((size_t)b * 4 + 3) * HWSZ + rem] = e3 * inv;
}

// ---------------------------------------------------------------------------
// Final: per pixel, softmax each flow's 25 kernel logits in registers, apply
// 5x5 adaptive conv over blended frames (zero pad), weight by cw, sum flows.
// kout: padded NHWC fp16, 128 channels; ch f*25+k holds logit k of flow f.
// ---------------------------------------------------------------------------
__global__ __launch_bounds__(256) void final_kernel(
    const _Float16* __restrict__ kout, const float* __restrict__ blended,
    const float* __restrict__ cw, float* __restrict__ outp)
{
    int pix = blockIdx.x * 256 + threadIdx.x;
    if (pix >= PIX) return;
    int b = pix / HWSZ, rem = pix % HWSZ;
    int y = rem / WW, x = rem % WW;
    float o0 = 0.f, o1 = 0.f, o2 = 0.f;
    for (int f = 0; f < 4; ++f) {
        const _Float16* kp = kout + pad_off(b, rem) * 128 + f * 25;
        float kv[25];
        float mx = -1e30f;
        #pragma unroll
        for (int k = 0; k < 25; ++k) { kv[k] = (float)kp[k]; mx = fmaxf(mx, kv[k]); }
        float s = 0.f;
        #pragma unroll
        for (int k = 0; k < 25; ++k) { kv[k] = __expf(kv[k] - mx); s += kv[k]; }
        float cwv = cw[((size_t)b * 4 + f) * HWSZ + rem];
        const float* blf = blended + (size_t)(f * NB + b) * 3 * HWSZ;
        float a0 = 0.f, a1 = 0.f, a2 = 0.f;
        #pragma unroll
        for (int k = 0; k < 25; ++k) {
            int yy = y + k / 5 - 2, xx = x + k % 5 - 2;
            if (yy >= 0 && yy < HH && xx >= 0 && xx < WW) {
                int r2 = yy * WW + xx;
                a0 += kv[k] * blf[r2];
                a1 += kv[k] * blf[HWSZ + r2];
                a2 += kv[k] * blf[2 * HWSZ + r2];
            }
        }
        float invs = cwv / s;
        o0 += a0 * invs; o1 += a1 * invs; o2 += a2 * invs;
    }
    outp[(size_t)b * 3 * HWSZ + rem]            = o0;
    outp[(size_t)b * 3 * HWSZ + HWSZ + rem]     = o1;
    outp[(size_t)b * 3 * HWSZ + 2 * HWSZ + rem] = o2;
}

// ---------------------------------------------------------------------------
// Host launcher. Input flattening follows jax tree order (dict keys sorted):
//  0 frame1, 1 frame2,
//  2..9   params["collab"]  : (w,b) x4
//  10..49 params["flow"][f] : head0 w,b  head1 w,b  head2 w,b  pyr0 w,b  pyr1 w,b
//  50..57 params["kernel"]  : (w,b) x4
// Output: out (B,3,H,W) fp32 then cw (B,4,H,W) fp32, concatenated.
// ---------------------------------------------------------------------------
extern "C" void kernel_launch(void* const* d_in, const int* in_sizes, int n_in,
                              void* d_out, int out_size, void* d_ws, size_t ws_size,
                              hipStream_t stream)
{
    (void)in_sizes; (void)n_in; (void)out_size; (void)ws_size;
    const float* frame1 = (const float*)d_in[0];
    const float* frame2 = (const float*)d_in[1];

    char*  ws  = (char*)d_ws;
    size_t off = 0;
    auto alloc = [&](size_t bytes) -> void* {
        void* p = ws + off;
        off += (bytes + 255) & ~(size_t)255;
        return p;
    };
    auto pack = [&](int widx, int Cout, int Cin, int CinSrc, int dup) -> const _Float16* {
        const float* w = (const float*)d_in[widx];
        int CoutPad = (Cout + 31) / 32 * 32, Mtiles = CoutPad / 16;
        int CinPad  = (Cin + 31) / 32 * 32,  nch   = CinPad / 32;
        int total = Mtiles * 9 * nch * 512;
        _Float16* wp = (_Float16*)alloc((size_t)total * 2);
        pack_w_kernel<<<(total + 255) / 256, 256, 0, stream>>>(
            w, wp, Cout, Cin, CinSrc, nch, dup, total);
        return wp;
    };
    auto conv = [&](const _Float16* act, const _Float16* wp, int bidx,
                    _Float16* outp, int Cin, int Cout, int relu, float scale) {
        const float* bias = (const float*)d_in[bidx];
        int CinPad  = (Cin + 31) / 32 * 32,  nch    = CinPad / 32;
        int CoutPad = (Cout + 31) / 32 * 32, Mpairs = CoutPad / 32;
        int tiles = Mpairs * (PIX / 32);
        conv3x3_wmma_kernel<<<(tiles + 7) / 8, 256, 0, stream>>>(
            act, wp, bias, outp, CinPad, nch, Cout, CoutPad, Mpairs, relu, scale);
    };

    // Padded activation buffers (lifetimes arranged for reuse).
    _Float16* x6      = (_Float16*)alloc((size_t)PIXP * 32 * 2);       // also x14
    _Float16* flows   = (_Float16*)alloc((size_t)PIXP * 32 * 2 * 4);   // 4 flows
    _Float16* featA   = (_Float16*)alloc((size_t)PIXP * 64 * 2);       // pyr0 / head0 out
    _Float16* featB   = (_Float16*)alloc((size_t)PIXP * 64 * 2);       // pyr1 out
    _Float16* featD   = (_Float16*)alloc((size_t)PIXP * 32 * 2);       // head1 / collab out
    _Float16* t128    = (_Float16*)alloc((size_t)PIXP * 128 * 2);      // also kernel logits
    _Float16* t256    = (_Float16*)alloc((size_t)PIXP * 256 * 2);
    _Float16* t512    = (_Float16*)alloc((size_t)PIXP * 512 * 2);
    float*    blended = (float*)alloc((size_t)4 * NB * 3 * HWSZ * 4);  // (F,B,3,H,W)

    int pixBlocks  = (PIX + 255) / 256;
    int pixpBlocks = (PIXP + 255) / 256;

    // Zero halos once per launch (interiors are fully written by convs).
    halo_zero_kernel<<<pixpBlocks, 256, 0, stream>>>(x6, 32);
    for (int f = 0; f < 4; ++f)
        halo_zero_kernel<<<pixpBlocks, 256, 0, stream>>>(flows + (size_t)f * PIXP * 32, 32);
    halo_zero_kernel<<<pixpBlocks, 256, 0, stream>>>(featA, 64);
    halo_zero_kernel<<<pixpBlocks, 256, 0, stream>>>(featB, 64);
    halo_zero_kernel<<<pixpBlocks, 256, 0, stream>>>(featD, 32);
    halo_zero_kernel<<<pixpBlocks, 256, 0, stream>>>(t128, 128);
    halo_zero_kernel<<<pixpBlocks, 256, 0, stream>>>(t256, 256);
    halo_zero_kernel<<<pixpBlocks, 256, 0, stream>>>(t512, 512);

    assemble_kernel<<<pixBlocks, 256, 0, stream>>>(frame1, frame2, nullptr, x6);

    // ---- 4 flow estimators ------------------------------------------------
    for (int f = 0; f < 4; ++f) {
        int ib = 10 + f * 10; // head0 w,b head1 w,b head2 w,b pyr0 w,b pyr1 w,b
        const _Float16* wp0 = pack(ib + 6, 64, 6,  6,   0);  // pyr0
        const _Float16* wp1 = pack(ib + 8, 64, 64, 64,  0);  // pyr1
        const _Float16* wh0 = pack(ib + 0, 64, 64, 128, 1);  // head0 (concat(f,f) folded)
        const _Float16* wh1 = pack(ib + 2, 32, 64, 64,  0);  // head1
        const _Float16* wh2 = pack(ib + 4, 2,  32, 32,  0);  // head2
        conv(x6,    wp0, ib + 7, featA, 6,  64, 1, 1.0f);
        conv(featA, wp1, ib + 9, featB, 64, 64, 1, 1.0f);
        conv(featB, wh0, ib + 1, featA, 64, 64, 1, 1.0f);
        conv(featA, wh1, ib + 3, featD, 64, 32, 1, 1.0f);
        conv(featD, wh2, ib + 5, flows + (size_t)f * PIXP * 32, 32, 2, 0, 0.5f); // *T
    }

    warp_blend_kernel<<<(4 * PIX + 255) / 256, 256, 0, stream>>>(
        frame1, frame2, flows, blended);
    assemble_kernel<<<pixBlocks, 256, 0, stream>>>(frame1, frame2, flows, x6); // x14

    float* out_img = (float*)d_out;                     // (B,3,H,W)
    float* out_cw  = out_img + (size_t)NB * 3 * HWSZ;   // (B,4,H,W)

    // ---- collaboration net ------------------------------------------------
    {
        const _Float16* w0 = pack(2, 128, 14,  14,  0);
        const _Float16* w1 = pack(4, 256, 128, 128, 0);
        const _Float16* w2 = pack(6, 128, 256, 256, 0);
        const _Float16* w3 = pack(8, 4,   128, 128, 0);
        conv(x6,   w0, 3, t128,  14,  128, 1, 1.0f);
        conv(t128, w1, 5, t256,  128, 256, 1, 1.0f);
        conv(t256, w2, 7, t128,  256, 128, 1, 1.0f);
        conv(t128, w3, 9, featD, 128, 4,   0, 1.0f);
    }
    cw_softmax_kernel<<<pixBlocks, 256, 0, stream>>>(featD, out_cw);

    // ---- kernel-prediction net -------------------------------------------
    {
        const _Float16* w0 = pack(50, 128, 14,  14,  0);
        const _Float16* w1 = pack(52, 256, 128, 128, 0);
        const _Float16* w2 = pack(54, 512, 256, 256, 0);
        const _Float16* w3 = pack(56, 100, 512, 512, 0);
        conv(x6,   w0, 51, t128, 14,  128, 1, 1.0f);
        conv(t128, w1, 53, t256, 128, 256, 1, 1.0f);
        conv(t256, w2, 55, t512, 256, 512, 1, 1.0f);
        conv(t512, w3, 57, t128, 512, 100, 0, 1.0f);   // kernel logits, Cpad=128
    }

    final_kernel<<<pixBlocks, 256, 0, stream>>>(t128, blended, out_cw, out_img);
}
